// CFRNet_48747878810213
// MI455X (gfx1250) — matrix operand order repaired and checked by reading, not compiled, not run on hardware
//
#include <hip/hip_runtime.h>
#include <hip/hip_bf16.h>
#include <cstdint>
#include <cstddef>

// ---------------------------------------------------------------------------
// CFRNet fused forward for gfx1250 (MI455X).
//  - All GEMMs via v_wmma_f32_16x16x32_bf16 (f32 accumulate).
//  - One block processes 64 rows through ALL layers with activations resident
//    in LDS (2 x 128KB ping-pong panels inside the 320KB WGP LDS).
//  - Input tile staged via GLOBAL_LOAD_ASYNC_TO_LDS_B128 (ASYNCcnt path).
//  - Weights pre-packed to bf16 in WMMA-fragment-major layout in d_ws so the
//    B operand is two contiguous b128 loads per lane per K-step.
//  - Two n-tiles per wave pass: each A fragment feeds 2 WMMAs (1 ds load per
//    wmma instead of 2).
// ---------------------------------------------------------------------------

typedef __attribute__((ext_vector_type(16))) __bf16   v16bf;
typedef __attribute__((ext_vector_type(8)))  float    v8f;
typedef __attribute__((ext_vector_type(4)))  unsigned u32x4;
typedef __attribute__((ext_vector_type(2)))  float    f32x2;
typedef __attribute__((ext_vector_type(4)))  float    f32x4;

union FragAB { v16bf bf; u32x4 u[2]; };

static constexpr int RM    = 64;    // rows per block
static constexpr int WAVES = 8;     // 256 threads, wave32
static constexpr int TPB   = 256;

__device__ __forceinline__ unsigned short f2bf(float f) {
  unsigned u = __builtin_bit_cast(unsigned, f);
  u += 0x7FFFu + ((u >> 16) & 1u);            // round-to-nearest-even
  return (unsigned short)(u >> 16);
}
__device__ __forceinline__ unsigned pack2bf(float lo, float hi) {
  return (unsigned)f2bf(lo) | ((unsigned)f2bf(hi) << 16);
}
__device__ __forceinline__ float bf_lo(unsigned u) { return __builtin_bit_cast(float, u << 16); }
__device__ __forceinline__ float bf_hi(unsigned u) { return __builtin_bit_cast(float, u & 0xFFFF0000u); }

// ---------------------------------------------------------------------------
// Pre-pass: convert f32 weight matrix (K x N, row-major) into bf16 packed in
// WMMA B-fragment-major order:  idx = (((kb*(N/16)+nt)*32 + lane)*8 + v)
// where lane 0-15 holds K = kb*32 + 2v,2v+1 and lane 16-31 holds K += 16,
// column n = nt*16 + (lane&15)   (contiguous k = 16h + s order, same as the A loads).
// ---------------------------------------------------------------------------
__global__ __launch_bounds__(256) void pack_weights(const float* __restrict__ W,
                                                    unsigned* __restrict__ dst,
                                                    int K, int N) {
  int idx   = blockIdx.x * blockDim.x + threadIdx.x;
  int total = (K * N) >> 1;
  if (idx >= total) return;
  int v     = idx & 7;
  int lane  = (idx >> 3) & 31;
  int t     = idx >> 8;
  int ntcnt = N >> 4;
  int nt    = t % ntcnt;
  int kb    = t / ntcnt;
  int n     = (nt << 4) + (lane & 15);
  int k0    = (kb << 5) + (lane & 16) + (v << 1);
  dst[idx]  = pack2bf(W[(size_t)k0 * N + n], W[(size_t)(k0 + 1) * N + n]);
}

// ---------------------------------------------------------------------------
// One dense layer: LDS(bf16, RM x K row-major) @ Wfrag(bf16 packed) -> LDS out
// (bf16, RM x N row-major), optional bias / ReLU / f32 global store (for OUT).
// Each wave owns n-tile PAIRS {wave, wave+8, ...}; per pair: 4 M-tiles, the A
// fragment is loaded once and used by both n-tiles (2 wmma per ds-b128 pair).
// ---------------------------------------------------------------------------
__device__ __forceinline__ void mlp_layer(
    char* lds, unsigned inOff, unsigned outOff,
    const unsigned* __restrict__ Wfrag, const float* __restrict__ bias,
    int K, int N, bool relu,
    float* __restrict__ gOut, long rowBase)
{
  const int tid      = threadIdx.x;
  const int wave     = tid >> 5;
  const int lane     = tid & 31;
  const int lane15   = lane & 15;
  const int laneHalf = (lane >> 4) & 1;
  const unsigned*  ldsIn  = reinterpret_cast<const unsigned*>(lds + inOff);
  unsigned short*  ldsOut = reinterpret_cast<unsigned short*>(lds + outOff);

  const int ntiles = N >> 4;
  const int npairs = ntiles >> 1;
  const int ksteps = K >> 5;
  for (int np = wave; np < npairs; np += WAVES) {
    const int nt0 = np << 1;
    v8f z = {0.f, 0.f, 0.f, 0.f, 0.f, 0.f, 0.f, 0.f};
    v8f acc[2][4] = {{z, z, z, z}, {z, z, z, z}};
    for (int kb = 0; kb < ksteps; ++kb) {
      // B fragments for both n-tiles: 2 x 32 contiguous bytes/lane (L2-hot)
      const unsigned* bp = Wfrag + (((size_t)kb * ntiles + nt0) * 32 + lane) * 8;
      FragAB bF0, bF1;
      bF0.u[0] = *reinterpret_cast<const u32x4*>(bp);
      bF0.u[1] = *reinterpret_cast<const u32x4*>(bp + 4);
      bF1.u[0] = *reinterpret_cast<const u32x4*>(bp + 256);
      bF1.u[1] = *reinterpret_cast<const u32x4*>(bp + 260);
#pragma unroll
      for (int mt = 0; mt < 4; ++mt) {
        // A fragment (contiguous k = 16h + s, as in B): two b128 LDS loads, reused twice
        int row = (mt << 4) + lane15;
        const unsigned* ap = ldsIn + (((size_t)row * K + (kb << 5) + laneHalf * 16) >> 1);
        FragAB aF;
        aF.u[0] = *reinterpret_cast<const u32x4*>(ap);
        aF.u[1] = *reinterpret_cast<const u32x4*>(ap + 4);
        acc[0][mt] = __builtin_amdgcn_wmma_f32_16x16x32_bf16(
            false, aF.bf, false, bF0.bf, (short)0, acc[0][mt], false, false);
        acc[1][mt] = __builtin_amdgcn_wmma_f32_16x16x32_bf16(
            false, aF.bf, false, bF1.bf, (short)0, acc[1][mt], false, false);
      }
    }
    // Epilogue: bias + ReLU, bf16 back to LDS, optional f32 global store.
#pragma unroll
    for (int half = 0; half < 2; ++half) {
      int   col = ((nt0 + half) << 4) + lane15;
      float bv  = bias ? bias[col] : 0.f;
#pragma unroll
      for (int mt = 0; mt < 4; ++mt) {
#pragma unroll
        for (int v = 0; v < 8; ++v) {
          int   row = (mt << 4) + laneHalf * 8 + v;   // C/D layout: hi lanes M+=8
          float x   = acc[half][mt][v] + bv;
          if (relu) x = fmaxf(x, 0.f);
          if (gOut) gOut[(rowBase + row) * (long)N + col] = x;
          ldsOut[(size_t)row * N + col] = f2bf(x);
        }
      }
    }
  }
}

// 512-wide dot of one LDS bf16 row with an f32 global vector, wave-reduced.
__device__ __forceinline__ float row_dot512(const char* lds, unsigned off, int row,
                                            const float* __restrict__ w) {
  const int lane = threadIdx.x & 31;
  const unsigned* p = reinterpret_cast<const unsigned*>(lds + off) + row * 256 + lane * 8;
  u32x4 u0 = *reinterpret_cast<const u32x4*>(p);
  u32x4 u1 = *reinterpret_cast<const u32x4*>(p + 4);
  const f32x4* wp = reinterpret_cast<const f32x4*>(w + lane * 16);
  f32x4 w0 = wp[0], w1 = wp[1], w2 = wp[2], w3 = wp[3];
  float s = 0.f;
  s += bf_lo(u0.x) * w0.x + bf_hi(u0.x) * w0.y;
  s += bf_lo(u0.y) * w0.z + bf_hi(u0.y) * w0.w;
  s += bf_lo(u0.z) * w1.x + bf_hi(u0.z) * w1.y;
  s += bf_lo(u0.w) * w1.z + bf_hi(u0.w) * w1.w;
  s += bf_lo(u1.x) * w2.x + bf_hi(u1.x) * w2.y;
  s += bf_lo(u1.y) * w2.z + bf_hi(u1.y) * w2.w;
  s += bf_lo(u1.z) * w3.x + bf_hi(u1.z) * w3.y;
  s += bf_lo(u1.w) * w3.z + bf_hi(u1.w) * w3.w;
#pragma unroll
  for (int m = 16; m >= 1; m >>= 1) s += __shfl_xor(s, m, 32);
  return s;
}

// ---------------------------------------------------------------------------
// Fully fused network: one block = 64 rows, activations never leave LDS.
// LDS map: R0=[0,128K) R1=[128K,256K) panels; YARR=[256K,+256B).
// ---------------------------------------------------------------------------
__global__ __launch_bounds__(256) void cfrnet_fused(
    const float* __restrict__ X, const int* __restrict__ treat,
    const unsigned* __restrict__ W0f, const float* __restrict__ b0,
    const unsigned* __restrict__ W1f, const float* __restrict__ b1,
    const unsigned* __restrict__ W2f, const float* __restrict__ b2,
    const unsigned* __restrict__ A0f, const float* __restrict__ a0,
    const unsigned* __restrict__ A1f, const float* __restrict__ a1,
    const float* __restrict__ A2,
    const unsigned* __restrict__ C0f, const float* __restrict__ c0,
    const unsigned* __restrict__ C1f, const float* __restrict__ c1,
    const float* __restrict__ C2,
    float* __restrict__ yOut, float* __restrict__ outG)
{
  extern __shared__ char lds[];
  constexpr unsigned R0   = 0;
  constexpr unsigned R0h1 = 65536;     // second half of R0 (64x512 bf16)
  constexpr unsigned R1   = 131072;
  constexpr unsigned YARR = 262144;
  const int  tid     = threadIdx.x;
  const long rowBase = (long)blockIdx.x * RM;

  // Phase 0a: async-copy input tile (64 x 512 f32 = 128KB) into R1 staging.
  // Global tile and LDS region are both contiguous with identical strides,
  // so one per-lane base + immediate offsets covers it (ASYNCcnt tracked).
  {
    unsigned long long ga =
        (unsigned long long)(const void*)(X + rowBase * 512) + (unsigned)(tid * 16);
    unsigned la = (unsigned)(uintptr_t)(lds + R1) + (unsigned)(tid * 16);
#pragma unroll
    for (int i = 0; i < 32; ++i) {          // 256 thr * 16B * 32 = 128KB
      asm volatile("global_load_async_to_lds_b128 %0, %1, off offset:%2"
                   :: "v"(la), "v"(ga), "i"(i * 4096) : "memory");
    }
    asm volatile("s_wait_asynccnt 0x0" ::: "memory");
  }
  __syncthreads();

  // Phase 0b: convert staged f32 tile -> bf16 row-major panel in R0.
  {
    const f32x2* xs  = reinterpret_cast<const f32x2*>(lds + R1);
    unsigned*    dst = reinterpret_cast<unsigned*>(lds + R0);
#pragma unroll 4
    for (int i = 0; i < (RM * 512 / 2) / TPB; ++i) {   // 64 iterations
      int   p = i * TPB + tid;
      f32x2 v = xs[p];
      dst[p]  = pack2bf(v.x, v.y);
    }
  }
  __syncthreads();

  // Shared net
  mlp_layer(lds, R0, R1, W0f, b0,  512, 1024, true,  nullptr, rowBase);
  __syncthreads();
  mlp_layer(lds, R1, R0, W1f, b1, 1024, 1024, true,  nullptr, rowBase);
  __syncthreads();
  // OUT layer: bias, no ReLU, also stream f32 result to d_out
  mlp_layer(lds, R0, R1, W2f, b2, 1024,  512, false, outG,    rowBase);
  __syncthreads();

  // Expert 0 (treatment == 0)
  mlp_layer(lds, R1, R0,   A0f, a0, 512, 512, true, nullptr, rowBase);
  __syncthreads();
  mlp_layer(lds, R0, R0h1, A1f, a1, 512, 512, true, nullptr, rowBase);
  __syncthreads();
  {
    const int wave = tid >> 5, lane = tid & 31;
    float* yArr = reinterpret_cast<float*>(lds + YARR);
    for (int r = wave; r < RM; r += WAVES) {
      float s = row_dot512(lds, R0h1, r, A2);
      if (lane == 0) yArr[r] = s;
    }
  }
  __syncthreads();

  // Expert 1 (treatment == 1); OUT still live in R1 first half
  mlp_layer(lds, R1, R0,   C0f, c0, 512, 512, true, nullptr, rowBase);
  __syncthreads();
  mlp_layer(lds, R0, R0h1, C1f, c1, 512, 512, true, nullptr, rowBase);
  __syncthreads();
  {
    const int wave = tid >> 5, lane = tid & 31;
    const float* yArr = reinterpret_cast<const float*>(lds + YARR);
    for (int r = wave; r < RM; r += WAVES) {
      float s = row_dot512(lds, R0h1, r, C2);
      if (lane == 0) {
        long gr  = rowBase + r;
        yOut[gr] = (treat[gr] == 0) ? yArr[r] : s;
      }
    }
  }
}

// ---------------------------------------------------------------------------
extern "C" void kernel_launch(void* const* d_in, const int* in_sizes, int n_in,
                              void* d_out, int out_size, void* d_ws, size_t ws_size,
                              hipStream_t stream) {
  (void)in_sizes; (void)n_in; (void)out_size; (void)ws_size;

  const float* X     = (const float*)d_in[0];
  const int*   treat = (const int*)d_in[1];
  const float* W0 = (const float*)d_in[2];   const float* b0 = (const float*)d_in[3];
  const float* W1 = (const float*)d_in[4];   const float* b1 = (const float*)d_in[5];
  const float* W2 = (const float*)d_in[6];   const float* b2 = (const float*)d_in[7];
  const float* A0 = (const float*)d_in[8];   const float* a0 = (const float*)d_in[9];
  const float* A1 = (const float*)d_in[10];  const float* a1 = (const float*)d_in[11];
  const float* A2 = (const float*)d_in[12];
  const float* C0 = (const float*)d_in[13];  const float* c0 = (const float*)d_in[14];
  const float* C1 = (const float*)d_in[15];  const float* c1 = (const float*)d_in[16];
  const float* C2 = (const float*)d_in[17];

  // bf16 packed weights in d_ws (u32 units); total ~6 MB.
  unsigned* ws  = (unsigned*)d_ws;
  unsigned* W0f = ws;                  // 512*1024/2 = 262144
  unsigned* W1f = W0f + 262144;        // 1024*1024/2 = 524288
  unsigned* W2f = W1f + 524288;        // 1024*512/2 = 262144
  unsigned* A0f = W2f + 262144;        // 512*512/2  = 131072
  unsigned* A1f = A0f + 131072;
  unsigned* C0f = A1f + 131072;
  unsigned* C1f = C0f + 131072;

  auto packLaunch = [&](const float* W, unsigned* dst, int K, int N) {
    int total = K * N / 2;
    pack_weights<<<(total + 255) / 256, 256, 0, stream>>>(W, dst, K, N);
  };
  packLaunch(W0, W0f,  512, 1024);
  packLaunch(W1, W1f, 1024, 1024);
  packLaunch(W2, W2f, 1024,  512);
  packLaunch(A0, A0f,  512,  512);
  packLaunch(A1, A1f,  512,  512);
  packLaunch(C0, C0f,  512,  512);
  packLaunch(C1, C1f,  512,  512);

  float* y    = (float*)d_out;          // 65536 floats
  float* outG = y + 65536;              // 65536*512 floats

  constexpr unsigned SMEM = 262144 + 256;   // 2x128KB panels + y0 scratch
  hipFuncSetAttribute(reinterpret_cast<const void*>(cfrnet_fused),
                      hipFuncAttributeMaxDynamicSharedMemorySize, (int)SMEM);
  cfrnet_fused<<<65536 / RM, TPB, SMEM, stream>>>(
      X, treat, W0f, b0, W1f, b1, W2f, b2,
      A0f, a0, A1f, a1, A2, C0f, c0, C1f, c1, C2, y, outG);
}
